// LocationAwareAttention_72224170049708
// MI455X (gfx1250) — matrix operand order repaired and checked by reading, not compiled, not run on hardware
//
#include <hip/hip_runtime.h>
#include <hip/hip_bf16.h>

// Problem constants (from reference)
#define B_  32
#define T_  2048
#define F_  1024
#define H_  16
#define D_  64
#define CO_ 10

typedef __attribute__((ext_vector_type(16))) __bf16 v16bf;
typedef __attribute__((ext_vector_type(8)))  float  v8f;
typedef unsigned int v4u __attribute__((ext_vector_type(4)));
typedef unsigned int v8u __attribute__((ext_vector_type(8)));

__device__ __forceinline__ unsigned short f2bf(float f) {
    unsigned u = __builtin_bit_cast(unsigned, f);
    unsigned r = ((u >> 16) & 1u) + 0x7FFFu;   // round-to-nearest-even
    return (unsigned short)((u + r) >> 16);
}
__device__ __forceinline__ float bf2f(unsigned short h) {
    unsigned u = ((unsigned)h) << 16;
    return __builtin_bit_cast(float, u);
}
__device__ __forceinline__ unsigned pack2(float a, float b) {
#if __has_builtin(__builtin_amdgcn_cvt_pk_bf16_f32)
    auto p = __builtin_amdgcn_cvt_pk_bf16_f32(a, b);   // 1 VALU op
    return __builtin_bit_cast(unsigned, p);
#else
    return (unsigned)f2bf(a) | ((unsigned)f2bf(b) << 16);
#endif
}
__device__ __forceinline__ void wait_tensorcnt0() {
#if __has_builtin(__builtin_amdgcn_s_wait_tensorcnt)
    __builtin_amdgcn_s_wait_tensorcnt(0);
#else
    asm volatile("s_wait_tensorcnt 0x0" ::: "memory");
#endif
}

// ---------------------------------------------------------------------------
// TDM 2D tile load: rows x 32 bf16 elements, row stride 1024 elements, into
// LDS with padded row stride 40 ushorts (pad_interval = 16 DWORDs data,
// pad_amount = 4 DWORDs).  D# per CDNA5 ISA.
// ---------------------------------------------------------------------------
__device__ __forceinline__ void tdm_load_2d(unsigned lds_off,
                                            unsigned long long gaddr,
                                            unsigned rows) {
    v4u g0;
    g0[0] = 1u;                                     // count=1, user descriptor
    g0[1] = lds_off;                                // lds_addr (bytes)
    g0[2] = (unsigned)gaddr;                        // global_addr[31:0]
    g0[3] = (unsigned)(gaddr >> 32) | (2u << 30);   // global_addr[56:32] | type=2
    v8u g1;
    // workgroup_mask=0 | data_size=1(2B)<<16 | pad_enable<<20 |
    // pad_interval=3(16 DW)<<22 | pad_amount=3(4 DW)<<25
    g1[0] = (1u << 16) | (1u << 20) | (3u << 22) | (3u << 25);
    g1[1] = (1024u << 16);        // tensor_dim0 = 1024 (bits 79:48, low half)
    g1[2] = (rows << 16);         // tensor_dim0 hi=0 | tensor_dim1 = rows (low half)
    g1[3] = (32u << 16);          // tensor_dim1 hi=0 | tile_dim0 = 32
    g1[4] = rows;                 // tile_dim1 = rows, tile_dim2 = 0
    g1[5] = 1024u;                // tensor_dim0_stride = 1024
    g1[6] = 0u;
    g1[7] = 0u;
    asm volatile("tensor_load_to_lds %0, %1" :: "s"(g0), "s"(g1) : "memory");
}

// ---------------------------------------------------------------------------
// Kernel 0: f32 -> packed bf16 (16 elements / thread), exact-multiple sizes
// ---------------------------------------------------------------------------
__global__ void to_bf16_kernel(const float* __restrict__ src,
                               uint4* __restrict__ dst) {
    size_t i = (size_t)blockIdx.x * 256 + threadIdx.x;
    const float4* s = (const float4*)src + i * 4;
    float4 a = s[0], b = s[1], c = s[2], d = s[3];
    uint4 o0, o1;
    o0.x = pack2(a.x, a.y); o0.y = pack2(a.z, a.w);
    o0.z = pack2(b.x, b.y); o0.w = pack2(b.z, b.w);
    o1.x = pack2(c.x, c.y); o1.y = pack2(c.z, c.w);
    o1.z = pack2(d.x, d.y); o1.w = pack2(d.z, d.w);
    dst[i * 2] = o0; dst[i * 2 + 1] = o1;
}

// ---------------------------------------------------------------------------
// Kernel 1: qb[b, h*64+d] = query[b,:] . Wq[h*64+d,:] + bias[d]   (bias folded)
// ---------------------------------------------------------------------------
__global__ void qproj_kernel(const float* __restrict__ query,
                             const float* __restrict__ Wq,
                             const float* __restrict__ bias,
                             float* __restrict__ qb) {
    int b = blockIdx.x;
    int o = blockIdx.y * 256 + threadIdx.x;
    const float4* qv = (const float4*)(query + (size_t)b * F_);
    const float4* wv = (const float4*)(Wq + (size_t)o * F_);
    float s = 0.f;
#pragma unroll 4
    for (int i = 0; i < F_ / 4; ++i) {
        float4 a = qv[i], w = wv[i];
        s += a.x * w.x + a.y * w.y + a.z * w.z + a.w * w.w;
    }
    qb[(size_t)b * F_ + o] = s + bias[o & (D_ - 1)];
}

// ---------------------------------------------------------------------------
// Kernel 2: loc[b,t,d] = tanh( sum_c (conv1d(prev_align)+cb)[b,c,t] * Wu[d,c] )
// ---------------------------------------------------------------------------
__global__ void loc_kernel(const float* __restrict__ prev_align,
                           const float* __restrict__ conv_w,
                           const float* __restrict__ conv_b,
                           const float* __restrict__ Wu,
                           float* __restrict__ loc) {
    __shared__ float pa[H_][66];
    __shared__ float cw[CO_ * H_ * 3];
    __shared__ float cb[CO_];
    __shared__ float wu[D_ * CO_];
    __shared__ float cbuf[CO_][64];

    int b = blockIdx.x, t0 = blockIdx.y * 64, tid = threadIdx.x;

    for (int i = tid; i < H_ * 66; i += 256) {
        int h = i / 66, tt = i % 66, gt = t0 + tt - 1;
        pa[h][tt] = (gt >= 0 && gt < T_) ? prev_align[((size_t)b * H_ + h) * T_ + gt] : 0.f;
    }
    for (int i = tid; i < CO_ * H_ * 3; i += 256) cw[i] = conv_w[i];
    if (tid < CO_) cb[tid] = conv_b[tid];
    for (int i = tid; i < D_ * CO_; i += 256) wu[i] = Wu[i];
    __syncthreads();

    for (int i = tid; i < CO_ * 64; i += 256) {
        int c = i / 64, tt = i % 64;
        float s = cb[c];
#pragma unroll
        for (int h = 0; h < H_; ++h) {
            const float* w3 = &cw[(c * H_ + h) * 3];
            s += pa[h][tt] * w3[0] + pa[h][tt + 1] * w3[1] + pa[h][tt + 2] * w3[2];
        }
        cbuf[c][tt] = s;
    }
    __syncthreads();

    for (int i = tid; i < 64 * D_; i += 256) {
        int tt = i >> 6, d = i & 63;
        float s = 0.f;
#pragma unroll
        for (int c = 0; c < CO_; ++c) s += cbuf[c][tt] * wu[d * CO_ + c];
        loc[((size_t)b * T_ + t0 + tt) * D_ + d] = tanhf(s);
    }
}

// ---------------------------------------------------------------------------
// Kernel 3 (dominant): TDM double-buffered bf16 WMMA GEMM + fused energy/score
//   v[256x64] = value[b, t0:t0+256, :] @ Wv[h*64:(h+1)*64, :]^T
// 8 waves x (32 rows x 64 cols) per block: 8 WMMA per 12 ds_load per K-step.
// ---------------------------------------------------------------------------
#define MT    256
#define KSTEP 32
#define ASTR  40     // ushort row stride (64B data + 16B pad from TDM)

__global__ void vproj_energy_kernel(const unsigned short* __restrict__ value_bf,
                                    const unsigned short* __restrict__ wv_bf,
                                    const float* __restrict__ qb,
                                    const float* __restrict__ loc,
                                    const float* __restrict__ score_w,
                                    const float* __restrict__ score_b,
                                    float* __restrict__ score,
                                    unsigned short* __restrict__ vws) {
    __shared__ __align__(16) unsigned short As[2][MT * ASTR];   // 40 KB
    __shared__ __align__(16) unsigned short Bs[2][64 * ASTR];   // 10 KB
    __shared__ float vstage[MT][D_ + 2];                        // 66 KB
    __shared__ float qh[D_], sw[D_];

    int t0 = blockIdx.x * MT;
    int h  = blockIdx.y;
    int b  = blockIdx.z;
    int tid  = threadIdx.x;
    int lane = tid & 31, wave = tid >> 5;
    int lrow = lane & 15, hi = lane >> 4;
    int m0 = wave * 32;                          // 8 waves x 32 rows

    if (tid < D_)          qh[tid]      = qb[(size_t)b * F_ + h * D_ + tid];
    else if (tid < 2 * D_) sw[tid - D_] = score_w[tid - D_];

    unsigned ldsA[2] = { (unsigned)(size_t)&As[0][0], (unsigned)(size_t)&As[1][0] };
    unsigned ldsB[2] = { (unsigned)(size_t)&Bs[0][0], (unsigned)(size_t)&Bs[1][0] };
    unsigned long long abase =
        (unsigned long long)(value_bf + ((size_t)b * T_ + t0) * F_);
    unsigned long long bbase =
        (unsigned long long)(wv_bf + (size_t)h * D_ * F_);

    // prologue: K-chunk 0 via TDM into buffer 0 (one wave issues the DMA)
    if (wave == 0) {
        tdm_load_2d(ldsA[0], abase, MT);
        tdm_load_2d(ldsB[0], bbase, 64);
        wait_tensorcnt0();
    }
    __syncthreads();

    union Frag { v16bf v; uint4 q[2]; };
    v8f acc0[4] = {}, acc1[4] = {};
    for (int kk = 0; kk < F_ / KSTEP; ++kk) {
        int cur = kk & 1;
        if (kk + 1 < F_ / KSTEP && wave == 0) {     // TDM prefetch next chunk
            unsigned long long koff = (unsigned long long)(kk + 1) * KSTEP * 2;
            tdm_load_2d(ldsA[cur ^ 1], abase + koff, MT);
            tdm_load_2d(ldsB[cur ^ 1], bbase + koff, 64);
        }
        // A fragments: lane<16 holds K{0..7,16..23}, lane>=16 K{8..15,24..31}
        Frag A0, A1, Bf[4];
        const uint4* a0p = (const uint4*)&As[cur][(m0 + lrow) * ASTR];
        const uint4* a1p = (const uint4*)&As[cur][(m0 + 16 + lrow) * ASTR];
        A0.q[0] = a0p[hi];     A0.q[1] = a0p[2 + hi];
        A1.q[0] = a1p[hi];     A1.q[1] = a1p[2 + hi];
#pragma unroll
        for (int j = 0; j < 4; ++j) {               // all B frags first
            const uint4* bp4 = (const uint4*)&Bs[cur][(j * 16 + lrow) * ASTR];
            Bf[j].q[0] = bp4[hi];
            Bf[j].q[1] = bp4[2 + hi];
        }
#pragma unroll
        for (int j = 0; j < 4; ++j) {               // 8 back-to-back WMMAs
            acc0[j] = __builtin_amdgcn_wmma_f32_16x16x32_bf16(
                false, A0.v, false, Bf[j].v, (short)0, acc0[j], false, false);
            acc1[j] = __builtin_amdgcn_wmma_f32_16x16x32_bf16(
                false, A1.v, false, Bf[j].v, (short)0, acc1[j], false, false);
        }
        if (kk + 1 < F_ / KSTEP) {
            if (wave == 0) wait_tensorcnt0();
            __syncthreads();
        }
    }

    // C layout: VGPR r -> row (+ r + 8*hi), col j*16 + lrow
#pragma unroll
    for (int j = 0; j < 4; ++j)
#pragma unroll
        for (int r = 0; r < 8; ++r) {
            vstage[m0 + r + 8 * hi][j * 16 + lrow]      = acc0[j][r];
            vstage[m0 + 16 + r + 8 * hi][j * 16 + lrow] = acc1[j][r];
        }
    __syncthreads();

    // emit v tile as packed bf16 (L2-resident for kernel 4)
    unsigned* vw32 = (unsigned*)vws;
    size_t vtp = (((size_t)(b * H_ + h) * T_ + t0) * D_) >> 1;
    for (int p = tid; p < MT * 32; p += 256) {
        int r = p >> 5, d2 = (p & 31) * 2;
        vw32[vtp + (size_t)r * 32 + (p & 31)] = pack2(vstage[r][d2], vstage[r][d2 + 1]);
    }

    // fused additive energy + score: one thread per row, no reduction needed
    const float* lp = loc + ((size_t)b * T_ + t0 + tid) * D_;
    float s = 0.f;
#pragma unroll 4
    for (int d = 0; d < D_; ++d) {
        float e = tanhf(vstage[tid][d] + qh[d] + lp[d]);    // bias folded into qh
        s += e * sw[d];
    }
    score[(size_t)(b * H_ + h) * T_ + t0 + tid] = s + score_b[0];
}

// ---------------------------------------------------------------------------
// Kernel 4: per (b,h): softmax over T, write align, then context = align . v
// ---------------------------------------------------------------------------
__global__ void softmax_context_kernel(const float* __restrict__ score,
                                       const unsigned short* __restrict__ vws,
                                       float* __restrict__ context,
                                       float* __restrict__ align_out) {
    __shared__ float sh[T_];
    __shared__ float red[256];
    int bh = blockIdx.x, tid = threadIdx.x;
    const float* sp = score + (size_t)bh * T_;

    float lv[8];
    float lmax = -3.4e38f;
#pragma unroll
    for (int i = 0; i < 8; ++i) {
        lv[i] = sp[tid + i * 256];
        lmax = fmaxf(lmax, lv[i]);
    }
    red[tid] = lmax; __syncthreads();
    for (int off = 128; off > 0; off >>= 1) {
        if (tid < off) red[tid] = fmaxf(red[tid], red[tid + off]);
        __syncthreads();
    }
    float m = red[0]; __syncthreads();

    float lsum = 0.f;
#pragma unroll
    for (int i = 0; i < 8; ++i) { lv[i] = expf(lv[i] - m); lsum += lv[i]; }
    red[tid] = lsum; __syncthreads();
    for (int off = 128; off > 0; off >>= 1) {
        if (tid < off) red[tid] += red[tid + off];
        __syncthreads();
    }
    float inv = 1.f / red[0]; __syncthreads();

#pragma unroll
    for (int i = 0; i < 8; ++i) {
        float a = lv[i] * inv;
        sh[tid + i * 256] = a;
        align_out[(size_t)bh * T_ + tid + i * 256] = a;
    }
    __syncthreads();

    int d = tid & 63, part = tid >> 6;
    const unsigned short* vp = vws + (size_t)bh * T_ * D_;
    float acc = 0.f;
    for (int t = part * 512; t < part * 512 + 512; ++t)
        acc += sh[t] * bf2f(vp[(size_t)t * D_ + d]);
    red[tid] = acc; __syncthreads();
    if (part == 0)
        context[(size_t)bh * D_ + d] = red[d] + red[64 + d] + red[128 + d] + red[192 + d];
}

// ---------------------------------------------------------------------------
extern "C" void kernel_launch(void* const* d_in, const int* in_sizes, int n_in,
                              void* d_out, int out_size, void* d_ws, size_t ws_size,
                              hipStream_t stream) {
    const float* query      = (const float*)d_in[0];
    const float* value      = (const float*)d_in[1];
    const float* prev_align = (const float*)d_in[2];
    const float* Wq         = (const float*)d_in[3];
    const float* Wv         = (const float*)d_in[4];
    const float* conv_w     = (const float*)d_in[5];
    const float* conv_b     = (const float*)d_in[6];
    const float* Wu         = (const float*)d_in[7];
    const float* bias       = (const float*)d_in[8];
    const float* score_w    = (const float*)d_in[9];
    const float* score_b    = (const float*)d_in[10];

    // workspace layout (bytes)
    char* ws = (char*)d_ws;
    float* qb    = (float*)(ws);                                  // 128 KB
    float* loc   = (float*)(ws + (128u << 10));                   // 16 MB
    float* score = (float*)(ws + (128u << 10) + (16u << 20));     // 4 MB
    unsigned short* vws =
        (unsigned short*)(ws + 21102592u);                        // 134 MB bf16 v
    unsigned short* value_bf =
        (unsigned short*)(ws + 155320320u);                       // 134 MB bf16 value
    unsigned short* wv_bf =
        (unsigned short*)(ws + 289538048u);                       // 2 MB bf16 Wv

    float* context   = (float*)d_out;                 // 32768 f32
    float* align_out = (float*)d_out + B_ * F_;       // 1048576 f32

    to_bf16_kernel<<<(B_ * T_ * F_) / (256 * 16), 256, 0, stream>>>(value, (uint4*)value_bf);
    to_bf16_kernel<<<(F_ * F_) / (256 * 16), 256, 0, stream>>>(Wv, (uint4*)wv_bf);
    qproj_kernel<<<dim3(B_, 4), 256, 0, stream>>>(query, Wq, bias, qb);
    loc_kernel<<<dim3(B_, T_ / 64), 256, 0, stream>>>(prev_align, conv_w, conv_b, Wu, loc);
    vproj_energy_kernel<<<dim3(T_ / MT, H_, B_), 256, 0, stream>>>(
        value_bf, wv_bf, qb, loc, score_w, score_b, score, vws);
    softmax_context_kernel<<<B_ * H_, 256, 0, stream>>>(score, vws, context, align_out);
}
